// ProbAttention_14130442403938
// MI455X (gfx1250) — matrix operand order repaired and checked by reading, not compiled, not run on hardware
//
#include <hip/hip_runtime.h>
#include <math.h>

// Problem constants (from reference: B=4, L=2048, H=8, D=64, FACTOR=5 -> u=U=n_top=40)
#define B_       4
#define L_       2048
#define H_       8
#define D_       64
#define S_       2048
#define U_SAMP   40
#define NTOP     40
#define NTOP_PAD 48      // 3 WMMA M-tiles of 16
#define CHUNK    32      // keys per streaming iteration (one WMMA K-dim for P*V)
#define NSLICE   8       // S-split factor for occupancy (256 attn blocks, 512 bcast blocks)
#define SLICE_S  (S_ / NSLICE)   // 256 keys per slice

typedef __attribute__((ext_vector_type(16))) __bf16 v16bf;
typedef __attribute__((ext_vector_type(8)))  float  v8f;

__device__ __forceinline__ v8f wmma_bf16(v16bf a, v16bf b, v8f c) {
  // D = A(16x32 bf16) * B(32x16 bf16) + C(16x16 f32)
  return __builtin_amdgcn_wmma_f32_16x16x32_bf16(
      /*neg_a=*/false, a, /*neg_b=*/false, b,
      /*c_mod=*/(short)0, c, /*reuse_a=*/false, /*reuse_b=*/false);
}

__device__ __forceinline__ unsigned short f32_bf16(float f) {
  union { float f; unsigned u; } v; v.f = f;
  unsigned u = v.u;
  u += 0x7FFFu + ((u >> 16) & 1u);   // round-to-nearest-even
  return (unsigned short)(u >> 16);
}

// Deterministic stand-in for the reference's fixed random sample indices
// (function of (l,u) only, shared across b,h as in the reference).
__device__ __forceinline__ unsigned samp_idx(unsigned l, unsigned u) {
  unsigned x = l * 40u + u;
  x ^= x >> 16; x *= 0x7feb352du;
  x ^= x >> 15; x *= 0x846ca68bu;
  x ^= x >> 16;
  return x & (S_ - 1u);
}

union U16 { uint4 q[2]; v16bf v; unsigned short s[16]; };

// ---------------------------------------------------------------------------
// Kernel 1: sparsity metric M[b,h,l] = max_u(q.k_samp) - sum_u(q.k_samp)/S
// One wave per (b,h,l); lanes split D=64 as 2 floats each.
// ---------------------------------------------------------------------------
__global__ void sample_score_kernel(const float* __restrict__ Q,
                                    const float* __restrict__ K,
                                    float* __restrict__ M) {
  int wave = (blockIdx.x * blockDim.x + threadIdx.x) >> 5;
  int lane = threadIdx.x & 31;
  int l  = wave & (L_ - 1);
  int bh = wave >> 11;                 // L_ == 2^11
  if (bh >= B_ * H_) return;

  const float* qrow  = Q + ((size_t)bh * L_ + l) * D_;
  const float* kbase = K + (size_t)bh * (size_t)S_ * D_;
  const float2 q2 = *(const float2*)(qrow + lane * 2);

  float mx = -__builtin_inff();
  float sm = 0.0f;
  for (int u = 0; u < U_SAMP; ++u) {
    unsigned idx = samp_idx((unsigned)l, (unsigned)u);
    const float2 k2 = *(const float2*)(kbase + (size_t)idx * D_ + lane * 2);
    float p = q2.x * k2.x + q2.y * k2.y;
    #pragma unroll
    for (int off = 16; off > 0; off >>= 1) p += __shfl_xor(p, off, 32);
    mx = fmaxf(mx, p);
    sm += p;
  }
  if (lane == 0) M[(size_t)bh * L_ + l] = mx - sm * (1.0f / (float)S_);
}

// ---------------------------------------------------------------------------
// Kernel 2: top-40 selection per (b,h) via 40 iterative argmax passes.
// One wave per (b,h).
// ---------------------------------------------------------------------------
__global__ void topk_kernel(const float* __restrict__ M, int* __restrict__ Mtop) {
  __shared__ float sm[L_];
  int bh   = blockIdx.x;
  int lane = threadIdx.x;
  const float* m = M + (size_t)bh * L_;
  for (int i = lane; i < L_; i += 32) sm[i] = m[i];
  __syncthreads();

  for (int t = 0; t < NTOP; ++t) {
    float best = -__builtin_inff();
    int   bidx = 0;
    for (int i = lane; i < L_; i += 32) {
      float v = sm[i];
      if (v > best) { best = v; bidx = i; }
    }
    #pragma unroll
    for (int off = 16; off > 0; off >>= 1) {
      float ov = __shfl_xor(best, off, 32);
      int   oi = __shfl_xor(bidx, off, 32);
      if (ov > best || (ov == best && oi < bidx)) { best = ov; bidx = oi; }
    }
    if (lane == 0) { Mtop[bh * NTOP + t] = bidx; sm[bidx] = -__builtin_inff(); }
    __syncthreads();
  }
}

// ---------------------------------------------------------------------------
// Kernel 3a: partial column sums of V over S-slices.
// Grid: (B*H)*NSLICE blocks; block (bh,s) sums rows [s*256, s*256+256).
// ---------------------------------------------------------------------------
__global__ void colsum_partial_kernel(const float* __restrict__ V,
                                      float* __restrict__ csum) {
  __shared__ float partial[256];
  int bh = blockIdx.x / NSLICE;
  int sl = blockIdx.x % NSLICE;
  int t  = threadIdx.x;
  const float* v = V + (size_t)bh * (size_t)S_ * D_ + (size_t)sl * SLICE_S * D_;

  int d = t & (D_ - 1), part = t >> 6;       // 4 partial sums per d column
  float s = 0.0f;
  for (int l = part; l < SLICE_S; l += 4) s += v[(size_t)l * D_ + d];
  partial[t] = s;
  __syncthreads();
  if (t < D_)
    csum[(size_t)blockIdx.x * D_ + t] =
        partial[t] + partial[t + 64] + partial[t + 128] + partial[t + 192];
}

// ---------------------------------------------------------------------------
// Kernel 3b: broadcast context = sum of slice colsums, written to all L rows.
// Grid: (B*H)*16 blocks; block (bh,c) writes rows [c*128, c*128+128).
// ---------------------------------------------------------------------------
__global__ void broadcast_kernel(const float* __restrict__ csum,
                                 float* __restrict__ out) {
  __shared__ float cs[D_];
  int bh = blockIdx.x >> 4;
  int lc = blockIdx.x & 15;
  int t  = threadIdx.x;
  if (t < D_) {
    float s = 0.0f;
    #pragma unroll
    for (int sl = 0; sl < NSLICE; ++sl)
      s += csum[((size_t)bh * NSLICE + sl) * D_ + t];
    cs[t] = s;
  }
  __syncthreads();
  float* o = out + (size_t)bh * (size_t)L_ * D_ + (size_t)lc * 128 * D_;
  for (int i = t; i < 128 * D_; i += 256) o[i] = cs[i & (D_ - 1)];
}

// ---------------------------------------------------------------------------
// Kernel 4a: partial flash attention on the 40 selected queries over one
// S-slice of 256 keys. Grid: (B*H)*NSLICE = 256 blocks; 3 waves per block,
// each owning one 16-row M-tile (48 padded rows). bf16 WMMA for QK^T and P*V.
// Emits UN-normalized accumulators + per-row (max, sumexp) stats.
// ---------------------------------------------------------------------------
__global__ void __launch_bounds__(96)
sparse_attn_partial_kernel(const float* __restrict__ Qg,
                           const float* __restrict__ Kg,
                           const float* __restrict__ Vg,
                           const int*   __restrict__ Mtop,
                           float*       __restrict__ pacc,
                           float*       __restrict__ pm,
                           float*       __restrict__ pl) {
  __shared__ unsigned short qlds[NTOP_PAD * D_];   // [m][d]   bf16, scale folded in
  __shared__ unsigned short klds[CHUNK * D_];      // [key][d] bf16
  __shared__ unsigned short vlds[D_ * CHUNK];      // [d][key] bf16 (transposed)
  __shared__ unsigned short plds[3][16 * CHUNK];   // per-wave P tile [m][key]
  __shared__ int            topidx[NTOP_PAD];

  const int bh   = blockIdx.x / NSLICE;
  const int sl   = blockIdx.x % NSLICE;
  const int tid  = threadIdx.x;       // 0..95
  const int wave = tid >> 5;          // 0..2  -> query rows [16w, 16w+16)
  const int lane = tid & 31;
  const int half = lane >> 4;         // lane group (0: lanes 0-15, 1: 16-31)
  const int ln16 = lane & 15;

  // ---- stage gathered top-query rows (pad rows alias row 0, never scattered) ----
  if (tid < NTOP_PAD) topidx[tid] = Mtop[bh * NTOP + (tid < NTOP ? tid : 0)];
  __syncthreads();
  for (int i = tid; i < NTOP_PAD * D_; i += 96) {
    int r = i >> 6, d = i & (D_ - 1);
    float q = Qg[((size_t)bh * L_ + topidx[r]) * D_ + d];
    qlds[i] = f32_bf16(q * 0.125f);   // fold scale = 1/sqrt(64)
  }
  __syncthreads();

  // ---- per-wave A-fragments of Q tile: 16x32 bf16, two d-chunks (0-31,32-63) ----
  // A k order chosen here (not AMD's table order): lane m=ln16; VGPRs hold K = half*16+[0..15]
  // contiguous (matches B).
  v16bf aQ[2];
  {
    const unsigned short* qrow = &qlds[(wave * 16 + ln16) * D_];
    #pragma unroll
    for (int ck = 0; ck < 2; ++ck) {
      U16 u16;
      int base = ck * 32 + half * 16;
      u16.q[0] = *(const uint4*)(qrow + base);        // 8 bf16 = 16B
      u16.q[1] = *(const uint4*)(qrow + base + 8);
      aQ[ck] = u16.v;
    }
  }

  // ---- online-softmax running state (rows replicated across 16-lane group) ----
  float m_i[8], l_i[8];
  v8f acc[4] = {v8f{}, v8f{}, v8f{}, v8f{}};
  #pragma unroll
  for (int i = 0; i < 8; ++i) { m_i[i] = -__builtin_inff(); l_i[i] = 0.0f; }

  const size_t kvbase = (size_t)bh * (size_t)S_ * D_;
  const int nchunks = SLICE_S / CHUNK;   // 8 chunks per slice

  for (int c = 0; c < nchunks; ++c) {
    __syncthreads();   // all waves done with previous chunk's LDS staging
    const int key0 = sl * SLICE_S + c * CHUNK;
    for (int i = tid; i < CHUNK * D_; i += 96) {
      int key = i >> 6, d = i & (D_ - 1);
      size_t g = kvbase + (size_t)(key0 + key) * D_ + d;
      klds[i] = f32_bf16(Kg[g]);                 // [key][d]
      vlds[d * CHUNK + key] = f32_bf16(Vg[g]);   // [d][key] transposed
    }
    if (c + 1 < nchunks) {                        // global_prefetch_b8
      __builtin_prefetch(Kg + kvbase + (size_t)(key0 + CHUNK) * D_, 0, 1);
      __builtin_prefetch(Vg + kvbase + (size_t)(key0 + CHUNK) * D_, 0, 1);
    }
    __syncthreads();

    // ---- scores: two 16x16 tiles (keys 0-15, 16-31), K-dim = D via 2 WMMAs ----
    // B k order chosen here (same as A, not AMD's table order): lane n=ln16; VGPRs hold K = half*16+[0..15]
    // contiguous -> contiguous 32B read from klds row.
    v8f s0 = {}, s1 = {};
    #pragma unroll
    for (int ck = 0; ck < 2; ++ck) {
      U16 b0, b1;
      const unsigned short* kr0 = &klds[(0 * 16 + ln16) * D_ + ck * 32 + half * 16];
      const unsigned short* kr1 = &klds[(1 * 16 + ln16) * D_ + ck * 32 + half * 16];
      b0.q[0] = *(const uint4*)(kr0); b0.q[1] = *(const uint4*)(kr0 + 8);
      b1.q[0] = *(const uint4*)(kr1); b1.q[1] = *(const uint4*)(kr1 + 8);
      s0 = wmma_bf16(aQ[ck], b0.v, s0);
      s1 = wmma_bf16(aQ[ck], b1.v, s1);
    }

    // ---- online softmax over this chunk's 32 columns ----
    // C layout: lane holds col n=ln16(+16); VGPR i = row half*8+i.
    float alpha[8];
    #pragma unroll
    for (int i = 0; i < 8; ++i) {
      float rm = fmaxf(s0[i], s1[i]);
      #pragma unroll
      for (int off = 1; off < 16; off <<= 1) rm = fmaxf(rm, __shfl_xor(rm, off, 16));
      float nm = fmaxf(m_i[i], rm);
      alpha[i] = __expf(m_i[i] - nm);
      m_i[i] = nm;
      float p0 = __expf(s0[i] - nm);
      float p1 = __expf(s1[i] - nm);
      float rs = p0 + p1;
      #pragma unroll
      for (int off = 1; off < 16; off <<= 1) rs += __shfl_xor(rs, off, 16);
      l_i[i] = l_i[i] * alpha[i] + rs;
      s0[i] = p0; s1[i] = p1;
    }

    // ---- transpose P (C layout -> A layout) through per-wave LDS tile ----
    {
      unsigned short* pw = &plds[wave][0];
      #pragma unroll
      for (int i = 0; i < 8; ++i) {
        int m = half * 8 + i;
        pw[m * CHUNK + ln16]      = f32_bf16(s0[i]);
        pw[m * CHUNK + 16 + ln16] = f32_bf16(s1[i]);
      }
    }
    asm volatile("s_wait_dscnt 0x0" ::: "memory");   // in-wave LDS store->load

    v16bf aP;
    {
      U16 u16;
      const unsigned short* pr = &plds[wave][ln16 * CHUNK];
      u16.q[0] = *(const uint4*)(pr + half * 16);
      u16.q[1] = *(const uint4*)(pr + half * 16 + 8);
      aP = u16.v;
    }

    // ---- rescale accumulators, then acc += P(16x32) * V(32x16) per d-tile ----
    #pragma unroll
    for (int t3 = 0; t3 < 4; ++t3) {
      #pragma unroll
      for (int i = 0; i < 8; ++i) acc[t3][i] *= alpha[i];
      U16 bv;
      const unsigned short* vr = &vlds[(t3 * 16 + ln16) * CHUNK + half * 16];
      bv.q[0] = *(const uint4*)(vr);
      bv.q[1] = *(const uint4*)(vr + 8);
      acc[t3] = wmma_bf16(aP, bv.v, acc[t3]);
    }
  }

  // ---- emit un-normalized partial accumulators + (max, sumexp) stats ----
  const size_t pbase = (size_t)blockIdx.x * NTOP_PAD;   // (bh, slice) partial row base
  #pragma unroll
  for (int i = 0; i < 8; ++i) {
    int r = wave * 16 + half * 8 + i;
    size_t arow = (pbase + r) * D_;
    pacc[arow + 0 * 16 + ln16] = acc[0][i];
    pacc[arow + 1 * 16 + ln16] = acc[1][i];
    pacc[arow + 2 * 16 + ln16] = acc[2][i];
    pacc[arow + 3 * 16 + ln16] = acc[3][i];
    if (ln16 == 0) { pm[pbase + r] = m_i[i]; pl[pbase + r] = l_i[i]; }
  }
}

// ---------------------------------------------------------------------------
// Kernel 4b: merge NSLICE partials (log-sum-exp merge) and scatter the 40
// rows into the broadcast context. Grid: (B*H)*NTOP blocks x 64 threads.
// ---------------------------------------------------------------------------
__global__ void combine_scatter_kernel(const float* __restrict__ pacc,
                                       const float* __restrict__ pm,
                                       const float* __restrict__ pl,
                                       const int*   __restrict__ Mtop,
                                       float*       __restrict__ out) {
  int blk = blockIdx.x;            // bh*NTOP + r
  int bh  = blk / NTOP;
  int r   = blk % NTOP;
  int d   = threadIdx.x;           // 0..63

  float gm = -__builtin_inff();
  #pragma unroll
  for (int sl = 0; sl < NSLICE; ++sl)
    gm = fmaxf(gm, pm[((size_t)bh * NSLICE + sl) * NTOP_PAD + r]);

  float den = 0.0f, num = 0.0f;
  #pragma unroll
  for (int sl = 0; sl < NSLICE; ++sl) {
    size_t idx = (size_t)bh * NSLICE + sl;
    float  w   = __expf(pm[idx * NTOP_PAD + r] - gm);
    den += pl[idx * NTOP_PAD + r] * w;
    num += pacc[(idx * NTOP_PAD + r) * D_ + d] * w;
  }
  out[((size_t)bh * L_ + Mtop[bh * NTOP + r]) * D_ + d] = num / den;
}

// ---------------------------------------------------------------------------
extern "C" void kernel_launch(void* const* d_in, const int* in_sizes, int n_in,
                              void* d_out, int out_size, void* d_ws, size_t ws_size,
                              hipStream_t stream) {
  (void)in_sizes; (void)n_in; (void)out_size; (void)ws_size;
  const float* Q = (const float*)d_in[0];   // (B,L,H,D) contiguous == (B,H,L,D) after reshape
  const float* K = (const float*)d_in[1];
  const float* V = (const float*)d_in[2];
  float* out = (float*)d_out;

  // Workspace layout (bytes): M | Mtop | colsum partials | pacc | pm | pl
  char* ws = (char*)d_ws;
  float* Mws  = (float*)ws;                                  ws += sizeof(float) * B_ * H_ * L_;
  int*   Mtop = (int*)ws;                                    ws += ((sizeof(int) * B_ * H_ * NTOP + 255) & ~255);
  float* csum = (float*)ws;                                  ws += sizeof(float) * B_ * H_ * NSLICE * D_;
  float* pacc = (float*)ws;                                  ws += sizeof(float) * B_ * H_ * NSLICE * NTOP_PAD * D_;
  float* pm   = (float*)ws;                                  ws += sizeof(float) * B_ * H_ * NSLICE * NTOP_PAD;
  float* pl   = (float*)ws;

  // 1) sparsity metric: one wave per (b,h,l), 8 waves/block (8192 blocks)
  sample_score_kernel<<<(B_ * H_ * L_) / 8, 256, 0, stream>>>(Q, K, Mws);
  // 2) top-40 per (b,h) (32 blocks, 1 wave each)
  topk_kernel<<<B_ * H_, 32, 0, stream>>>(Mws, Mtop);
  // 3) broadcast context: partial colsums (256 blocks) then parallel write (512 blocks)
  colsum_partial_kernel<<<B_ * H_ * NSLICE, 256, 0, stream>>>(V, csum);
  broadcast_kernel<<<B_ * H_ * 16, 256, 0, stream>>>(csum, out);
  // 4) WMMA sparse attention: S-split partials (256 blocks), then merge+scatter
  sparse_attn_partial_kernel<<<B_ * H_ * NSLICE, 96, 0, stream>>>(Q, K, V, Mtop, pacc, pm, pl);
  combine_scatter_kernel<<<B_ * H_ * NTOP, 64, 0, stream>>>(pacc, pm, pl, Mtop, out);
}
